// ShapeAttention_69020124446954
// MI455X (gfx1250) — matrix operand order repaired and checked by reading, not compiled, not run on hardware
//
#include <hip/hip_runtime.h>
#include <hip/hip_bf16.h>

// ---------------------------------------------------------------------------
// CDNA5 (gfx1250) ShapeAttention block.
//  - All GEMMs on v_wmma_f32_16x16x32_bf16 (bf16 operands, fp32 accum).
//  - A operands: row-major [M][K]. B operands: n-major [N][K] (k contiguous),
//    produced in that layout by the pack kernels -> every LDS stage is a
//    straight 16B copy, issued as GLOBAL_LOAD_ASYNC_TO_LDS_B128 with
//    double-buffered tiles and s_wait_asynccnt pipelining.
//  - wave32; 256-thread blocks = 8 waves (4x2); 256x128 block tile;
//    64x64 per-wave tile = 4x4 WMMA fragments -> 1 ds_load_b128 per WMMA.
// ---------------------------------------------------------------------------

typedef __bf16 bf16;
typedef __attribute__((ext_vector_type(16))) __bf16   v16bf;
typedef __attribute__((ext_vector_type(8)))  float    v8f;
typedef __attribute__((ext_vector_type(4)))  unsigned uv4;

union Frag {
    v16bf v;
    bf16  e[16];
    unsigned u[8];
    uv4 q[2];
};

#define TM 256
#define TN 128
#define TK 32

// async global -> LDS 16-byte copy (CDNA5, tracked by ASYNCcnt)
__device__ __forceinline__ void async_b128(void* lds_dst, const void* gsrc) {
    asm volatile("global_load_async_to_lds_b128 %0, %1, off"
                 :: "v"((unsigned)(uintptr_t)lds_dst), "v"(gsrc)
                 : "memory");
}
__device__ __forceinline__ void wait_async0() {
    asm volatile("s_wait_asynccnt 0" ::: "memory");
}

// ---------------------------------------------------------------------------
// Batched GEMM: D[b] = A[b] (MxK row-major) * B[b] ([N][K] n-major) + bias
// M % 256 == 0, N % 128 == 0, K % 32 == 0 at every call site.
// ---------------------------------------------------------------------------
__global__ __launch_bounds__(256)
void k_gemm_bf16(const bf16* __restrict__ A, const bf16* __restrict__ Bm,
                 const float* __restrict__ bias, float* __restrict__ D,
                 int M, int N, int K,
                 long long a_bstride, long long b_bstride, long long d_bstride,
                 long long ldb)
{
    const int b = blockIdx.z;
    A  += (size_t)a_bstride * b;
    Bm += (size_t)b_bstride * b;
    D  += (size_t)d_bstride * b;

    const int row0 = blockIdx.y * TM;
    const int n0   = blockIdx.x * TN;

    __shared__ bf16 As[2][TM][TK];   // 32 KB
    __shared__ bf16 Bs[2][TN][TK];   // 16 KB

    const int tid  = threadIdx.x;
    const int wave = tid >> 5;
    const int lane = tid & 31;
    const int half = lane >> 4;
    const int lrow = lane & 15;
    const int wm   = (wave & 3) * 64;   // wave row base inside 256
    const int wn   = (wave >> 2) * 64;  // wave col base inside 128

    // staging assignments (all 16B-aligned straight copies)
    const int ar = tid;                 // A row 0..255, full 32-elem row
    const int bn = tid >> 1;            // B row (n) 0..127
    const int bk = (tid & 1) << 4;      // B col 0 or 16

    v8f acc[4][4];
#pragma unroll
    for (int i = 0; i < 4; ++i)
#pragma unroll
        for (int j = 0; j < 4; ++j)
            acc[i][j] = (v8f){0, 0, 0, 0, 0, 0, 0, 0};

    const int nk = K / TK;

    // prologue: stage tile 0 into buffer 0
    {
        const bf16* asrc = A + (size_t)(row0 + ar) * K;
        async_b128(&As[0][ar][0],  asrc);
        async_b128(&As[0][ar][8],  asrc + 8);
        async_b128(&As[0][ar][16], asrc + 16);
        async_b128(&As[0][ar][24], asrc + 24);
        const bf16* bsrc = Bm + (size_t)(n0 + bn) * ldb + bk;
        async_b128(&Bs[0][bn][bk],     bsrc);
        async_b128(&Bs[0][bn][bk + 8], bsrc + 8);
    }
    wait_async0();
    __syncthreads();

    int buf = 0;
    for (int kb = 0; kb < nk; ++kb) {
        // issue async copies for the next tile into the other buffer
        if (kb + 1 < nk) {
            const int k0 = (kb + 1) * TK;
            const bf16* asrc = A + (size_t)(row0 + ar) * K + k0;
            async_b128(&As[buf ^ 1][ar][0],  asrc);
            async_b128(&As[buf ^ 1][ar][8],  asrc + 8);
            async_b128(&As[buf ^ 1][ar][16], asrc + 16);
            async_b128(&As[buf ^ 1][ar][24], asrc + 24);
            const bf16* bsrc = Bm + (size_t)(n0 + bn) * ldb + k0 + bk;
            async_b128(&Bs[buf ^ 1][bn][bk],     bsrc);
            async_b128(&Bs[buf ^ 1][bn][bk + 8], bsrc + 8);
        }

        // ---- fragments (ISA 7.12.2 bf16 layouts), all b128 LDS reads
        Frag af[4], bfr[4];
#pragma unroll
        for (int tm = 0; tm < 4; ++tm) {
            const uv4* ap = (const uv4*)&As[buf][wm + tm * 16 + lrow][0];
            af[tm].q[0] = ap[half];
            af[tm].q[1] = ap[2 + half];
        }
#pragma unroll
        for (int tn = 0; tn < 4; ++tn) {
            const uv4* bp = (const uv4*)&Bs[buf][wn + tn * 16 + lrow][0];
            bfr[tn].q[0] = bp[half];
            bfr[tn].q[1] = bp[2 + half];
        }
#pragma unroll
        for (int tm = 0; tm < 4; ++tm)
#pragma unroll
            for (int tn = 0; tn < 4; ++tn)
                acc[tm][tn] = __builtin_amdgcn_wmma_f32_16x16x32_bf16(
                    false, af[tm].v, false, bfr[tn].v,
                    (short)0, acc[tm][tn], false, false);

        wait_async0();       // next tile landed in LDS
        __syncthreads();     // everyone done reading current buffer
        buf ^= 1;
    }

    // ---- epilogue: bias + fp32 store (D layout: VGPR v -> M=v+8*half, N=lrow)
#pragma unroll
    for (int tm = 0; tm < 4; ++tm) {
#pragma unroll
        for (int tn = 0; tn < 4; ++tn) {
            const int col = n0 + wn + tn * 16 + lrow;
#pragma unroll
            for (int v = 0; v < 8; ++v) {
                const int m = row0 + wm + tm * 16 + v + half * 8;
                float r = acc[tm][tn][v];
                if (bias) r += bias[m];
                D[(size_t)m * N + col] = r;
            }
        }
    }
}

// ---------------------------------------------------------------------------
// Elementwise / pack kernels
// ---------------------------------------------------------------------------
__device__ __forceinline__ float silu_f(float v) { return v / (1.f + __expf(-v)); }

__global__ void k_cvt_bf16(const float* __restrict__ src, bf16* __restrict__ dst,
                           long long n) {
    long long i = (long long)blockIdx.x * blockDim.x + threadIdx.x;
    if (i < n) dst[i] = (bf16)src[i];
}

// dst [B][N][C] (B-operand layout) = cvt(src [B][C][N])
__global__ void k_packT_cvt(const float* __restrict__ src, bf16* __restrict__ dst,
                            int C, int N, long long total) {
    long long i = (long long)blockIdx.x * blockDim.x + threadIdx.x;
    if (i >= total) return;
    int c = (int)(i % C);
    long long r = i / C;
    int n = (int)(r % N);
    int b = (int)(r / N);
    dst[i] = (bf16)src[((size_t)b * C + c) * N + n];
}

__global__ void k_packT_silu(const float* __restrict__ src, bf16* __restrict__ dst,
                             int C, int N, long long total) {
    long long i = (long long)blockIdx.x * blockDim.x + threadIdx.x;
    if (i >= total) return;
    int c = (int)(i % C);
    long long r = i / C;
    int n = (int)(r % N);
    int b = (int)(r / N);
    dst[i] = (bf16)silu_f(src[((size_t)b * C + c) * N + n]);
}

__global__ void k_packT_relu(const float* __restrict__ src, bf16* __restrict__ dst,
                             int C, int N, long long total) {
    long long i = (long long)blockIdx.x * blockDim.x + threadIdx.x;
    if (i >= total) return;
    int c = (int)(i % C);
    long long r = i / C;
    int n = (int)(r % N);
    int b = (int)(r / N);
    dst[i] = (bf16)fmaxf(src[((size_t)b * C + c) * N + n], 0.f);
}

__global__ void k_packT_mulsilu(const float* __restrict__ a, const float* __restrict__ c2,
                                bf16* __restrict__ dst, int C, int N, long long total) {
    long long i = (long long)blockIdx.x * blockDim.x + threadIdx.x;
    if (i >= total) return;
    int c = (int)(i % C);
    long long r = i / C;
    int n = (int)(r % N);
    int b = (int)(r / N);
    size_t s = ((size_t)b * C + c) * N + n;
    float v = a[s] * c2[s];
    dst[i] = (bf16)silu_f(v);
}

// dst[b,c,n] = bf16(src[b, chOff + c, n]);  src has Csrc channels ([B,C,N] layout)
__global__ void k_split_pack(const float* __restrict__ src, bf16* __restrict__ dst,
                             int C, int N, int chOff, int Csrc, long long total) {
    long long i = (long long)blockIdx.x * blockDim.x + threadIdx.x;
    if (i >= total) return;
    int n = (int)(i % N);
    long long r = i / N;
    int c = (int)(r % C);
    int b = (int)(r / C);
    dst[i] = (bf16)src[((size_t)b * Csrc + chOff + c) * N + n];
}

// per-channel biased mean / variance over (B, N) -- one block per channel
__global__ __launch_bounds__(256)
void k_bnstats(const float* __restrict__ x, float* __restrict__ mean,
               float* __restrict__ var, int Bn, int C, int N) {
    const int c = blockIdx.x;
    float s = 0.f, s2 = 0.f;
    for (long long i = threadIdx.x; i < (long long)Bn * N; i += blockDim.x) {
        int b = (int)(i / N);
        int n = (int)(i % N);
        float v = x[((size_t)b * C + c) * N + n];
        s += v; s2 += v * v;
    }
    __shared__ float rs[256], rq[256];
    rs[threadIdx.x] = s; rq[threadIdx.x] = s2;
    __syncthreads();
    for (int st = 128; st > 0; st >>= 1) {
        if ((int)threadIdx.x < st) {
            rs[threadIdx.x] += rs[threadIdx.x + st];
            rq[threadIdx.x] += rq[threadIdx.x + st];
        }
        __syncthreads();
    }
    if (threadIdx.x == 0) {
        float inv = 1.f / ((float)Bn * (float)N);
        float m = rs[0] * inv;
        mean[c] = m;
        var[c]  = rq[0] * inv - m * m;
    }
}

// out = BN(x)*(1+scale)+shift.  outf (optional): [B][C][N] f32.
// outb: bf16 in B-operand layout [B][N][C].
// mod is [B,3C,N]: gate [0,C), shift [C,2C), scale [2C,3C)
__global__ void k_modulate(const float* __restrict__ xsrc, const float* __restrict__ mod,
                           const float* __restrict__ mean, const float* __restrict__ var,
                           const float* __restrict__ gamma, const float* __restrict__ beta,
                           float* __restrict__ outf, bf16* __restrict__ outb,
                           int C, int N, long long total) {
    long long i = (long long)blockIdx.x * blockDim.x + threadIdx.x;
    if (i >= total) return;
    int n = (int)(i % N);
    long long r = i / N;
    int c = (int)(r % C);
    int b = (int)(r / C);
    size_t mbase = (size_t)b * 3 * C * N + n;
    float shift = mod[mbase + (size_t)(C + c) * N];
    float scale = mod[mbase + (size_t)(2 * C + c) * N];
    float xn = (xsrc[i] - mean[c]) * rsqrtf(var[c] + 1e-5f) * gamma[c] + beta[c];
    float o = xn * (1.f + scale) + shift;
    if (outf) outf[i] = o;
    outb[((size_t)b * N + n) * C + c] = (bf16)o;
}

// xo = x_e + gate * (x1 - x_shape)
__global__ void k_xo(const float* __restrict__ x_e, const float* __restrict__ mod,
                     const float* __restrict__ x1f, const float* __restrict__ xsh,
                     float* __restrict__ xo, int C, int N, long long total) {
    long long i = (long long)blockIdx.x * blockDim.x + threadIdx.x;
    if (i >= total) return;
    int n = (int)(i % N);
    long long r = i / N;
    int c = (int)(r % C);
    int b = (int)(r / C);
    float gate = mod[(size_t)b * 3 * C * N + (size_t)c * N + n];
    xo[i] = x_e[i] + gate * (x1f[i] - xsh[i]);
}

// out = shortcut + gate_l * mlp
__global__ void k_final(const float* __restrict__ x_e, const float* __restrict__ mod2,
                        const float* __restrict__ mlp, float* __restrict__ out,
                        int C, int N, long long total) {
    long long i = (long long)blockIdx.x * blockDim.x + threadIdx.x;
    if (i >= total) return;
    int n = (int)(i % N);
    long long r = i / N;
    int c = (int)(r % C);
    int b = (int)(r / C);
    float gate = mod2[(size_t)b * 3 * C * N + (size_t)c * N + n];
    out[i] = x_e[i] + gate * mlp[i];
}

// in-place row softmax over last dim; grid = (C rows, B)
__global__ __launch_bounds__(256)
void k_softmax_row(float* __restrict__ e, int C) {
    float* row = e + ((size_t)blockIdx.y * gridDim.x + blockIdx.x) * C;
    __shared__ float red[256];
    float mx = -3.4e38f;
    for (int j = threadIdx.x; j < C; j += 256) mx = fmaxf(mx, row[j]);
    red[threadIdx.x] = mx;
    __syncthreads();
    for (int st = 128; st > 0; st >>= 1) {
        if ((int)threadIdx.x < st)
            red[threadIdx.x] = fmaxf(red[threadIdx.x], red[threadIdx.x + st]);
        __syncthreads();
    }
    mx = red[0];
    __syncthreads();
    float sm = 0.f;
    for (int j = threadIdx.x; j < C; j += 256) sm += __expf(row[j] - mx);
    red[threadIdx.x] = sm;
    __syncthreads();
    for (int st = 128; st > 0; st >>= 1) {
        if ((int)threadIdx.x < st) red[threadIdx.x] += red[threadIdx.x + st];
        __syncthreads();
    }
    float inv = 1.f / red[0];
    __syncthreads();
    for (int j = threadIdx.x; j < C; j += 256) row[j] = __expf(row[j] - mx) * inv;
}

// colsum[b][j] = sum_i attn[b][i][j]
__global__ void k_colsum(const float* __restrict__ attn, float* __restrict__ cs, int C) {
    int b = blockIdx.y;
    int j = blockIdx.x * blockDim.x + threadIdx.x;
    if (j >= C) return;
    const float* p = attn + (size_t)b * C * C + j;
    float s = 0.f;
    for (int i2 = 0; i2 < C; ++i2) s += p[(size_t)i2 * C];
    cs[(size_t)b * C + j] = s;
}

// attn_bf = bf16( attn / (1e-9 + colsum[j]) )   (A-operand layout, row-major)
__global__ void k_attn_pack(const float* __restrict__ attn, const float* __restrict__ cs,
                            bf16* __restrict__ dst, int C, long long total) {
    long long i = (long long)blockIdx.x * blockDim.x + threadIdx.x;
    if (i >= total) return;
    int j = (int)(i % C);
    int b = (int)(i / ((long long)C * C));
    dst[i] = (bf16)(attn[i] / (1e-9f + cs[(size_t)b * C + j]));
}

// ---------------------------------------------------------------------------
// Host launcher
// ---------------------------------------------------------------------------
extern "C" void kernel_launch(void* const* d_in, const int* in_sizes, int n_in,
                              void* d_out, int out_size, void* d_ws, size_t ws_size,
                              hipStream_t stream) {
    (void)in_sizes; (void)n_in; (void)out_size; (void)ws_size;

    constexpr int  Bn = 16, Nn = 2048, C = 512, DIN = 128;
    const long long SZ  = (long long)Bn * C * Nn;        // 16,777,216
    const long long SZI = (long long)Bn * DIN * Nn;      // 4,194,304
    const long long SZE = (long long)Bn * C * C;         // 4,194,304

    const float* time_p  = (const float*)d_in[0];
    const float* x_p     = (const float*)d_in[1];
    const float* shape_p = (const float*)d_in[2];
    const float* W_x  = (const float*)d_in[3];  const float* b_x  = (const float*)d_in[4];
    const float* W_t  = (const float*)d_in[5];  const float* b_t  = (const float*)d_in[6];
    const float* W_s  = (const float*)d_in[7];  const float* b_s  = (const float*)d_in[8];
    const float* W_ada= (const float*)d_in[9];  const float* b_ada= (const float*)d_in[10];
    const float* W_kv = (const float*)d_in[11]; const float* b_kv = (const float*)d_in[12];
    const float* W_tt = (const float*)d_in[13]; const float* b_tt = (const float*)d_in[14];
    const float* W_m1 = (const float*)d_in[15]; const float* b_m1 = (const float*)d_in[16];
    const float* W_m2 = (const float*)d_in[17]; const float* b_m2 = (const float*)d_in[18];
    const float* gamma1 = (const float*)d_in[19]; const float* beta1 = (const float*)d_in[20];
    const float* gamma2 = (const float*)d_in[21]; const float* beta2 = (const float*)d_in[22];
    float* out = (float*)d_out;

    // bump allocator over d_ws (256B aligned)
    char* wsb = (char*)d_ws;
    size_t off = 0;
    auto alloc = [&](size_t bytes) -> void* {
        void* p = wsb + off;
        off = (off + bytes + 255) & ~(size_t)255;
        return p;
    };

    // bf16 weight copies (A operands, row-major [M][K])
    bf16* wx_bf  = (bf16*)alloc((size_t)512 * 128 * 2);
    bf16* wt_bf  = (bf16*)alloc((size_t)512 * 128 * 2);
    bf16* ws_bf  = (bf16*)alloc((size_t)512 * 128 * 2);
    bf16* wada_bf= (bf16*)alloc((size_t)1536 * 512 * 2);
    bf16* wkv_bf = (bf16*)alloc((size_t)1024 * 512 * 2);
    bf16* wtt_bf = (bf16*)alloc((size_t)1536 * 512 * 2);
    bf16* wm1_bf = (bf16*)alloc((size_t)1024 * 512 * 2);
    bf16* wm2_bf = (bf16*)alloc((size_t)512 * 1024 * 2);
    // bf16 activations (B operands: [batch][n][k], k contiguous)
    bf16* x_bf     = (bf16*)alloc((size_t)SZI * 2);
    bf16* time_bf  = (bf16*)alloc((size_t)SZI * 2);
    bf16* shape_bf = (bf16*)alloc((size_t)SZI * 2);
    bf16* ths_bf   = (bf16*)alloc((size_t)SZ * 2);
    bf16* rs_bf    = (bf16*)alloc((size_t)SZ * 2);
    bf16* st_bf    = (bf16*)alloc((size_t)SZ * 2);
    bf16* kk_bf    = (bf16*)alloc((size_t)SZ * 2);   // A operand [C][N]
    bf16* vv_bf    = (bf16*)alloc((size_t)SZ * 2);   // B operand [j][n] == [C][N]
    bf16* x1_bf    = (bf16*)alloc((size_t)SZ * 2);   // B operand [n][C]
    bf16* attn_bf  = (bf16*)alloc((size_t)SZE * 2);  // A operand [C][C]
    bf16* h_bf     = (bf16*)alloc((size_t)SZ * 2);   // B operand [n][C]
    bf16* m1s_bf   = (bf16*)alloc((size_t)2 * SZ * 2); // B operand [n][1024]
    // f32 intermediates
    float* x_e   = (float*)alloc((size_t)SZ * 4);
    float* t_f   = (float*)alloc((size_t)SZ * 4);
    float* s_f   = (float*)alloc((size_t)SZ * 4);
    float* mod   = (float*)alloc((size_t)3 * SZ * 4);
    float* mod2  = (float*)alloc((size_t)3 * SZ * 4);
    float* kv    = (float*)alloc((size_t)2 * SZ * 4);
    float* x1f   = (float*)alloc((size_t)SZ * 4);
    float* energy= (float*)alloc((size_t)SZE * 4);
    float* xshape= (float*)alloc((size_t)SZ * 4);
    float* xo    = (float*)alloc((size_t)SZ * 4);
    float* m1o   = (float*)alloc((size_t)2 * SZ * 4);
    float* mlp   = (float*)alloc((size_t)SZ * 4);
    float* colsum= (float*)alloc((size_t)Bn * C * 4);
    float* mean1 = (float*)alloc((size_t)C * 4);
    float* var1  = (float*)alloc((size_t)C * 4);
    float* mean2 = (float*)alloc((size_t)C * 4);
    float* var2  = (float*)alloc((size_t)C * 4);

    auto eb = [](long long n) { return dim3((unsigned)((n + 255) / 256)); };
    const dim3 tb(256);

    auto gemm = [&](const bf16* A, const bf16* Bp, const float* bias, float* Dp,
                    int M, int Nc, int K,
                    long long ab, long long bb, long long db, long long ldb) {
        dim3 g((unsigned)(Nc / TN), (unsigned)(M / TM), (unsigned)Bn);
        k_gemm_bf16<<<g, tb, 0, stream>>>(A, Bp, bias, Dp, M, Nc, K, ab, bb, db, ldb);
    };

    // ---- pack weights to bf16
    k_cvt_bf16<<<eb(512 * 128), tb, 0, stream>>>(W_x,  wx_bf,  512 * 128);
    k_cvt_bf16<<<eb(512 * 128), tb, 0, stream>>>(W_t,  wt_bf,  512 * 128);
    k_cvt_bf16<<<eb(512 * 128), tb, 0, stream>>>(W_s,  ws_bf,  512 * 128);
    k_cvt_bf16<<<eb(1536 * 512), tb, 0, stream>>>(W_ada, wada_bf, 1536 * 512);
    k_cvt_bf16<<<eb(1024 * 512), tb, 0, stream>>>(W_kv,  wkv_bf,  1024 * 512);
    k_cvt_bf16<<<eb(1536 * 512), tb, 0, stream>>>(W_tt,  wtt_bf,  1536 * 512);
    k_cvt_bf16<<<eb(1024 * 512), tb, 0, stream>>>(W_m1,  wm1_bf,  1024 * 512);
    k_cvt_bf16<<<eb(512 * 1024), tb, 0, stream>>>(W_m2,  wm2_bf,  512 * 1024);

    // ---- pack activations into B-operand layout [b][n][k]
    // time/shape are already [B,N,128] -> direct convert; x needs transpose
    k_cvt_bf16<<<eb(SZI), tb, 0, stream>>>(time_p,  time_bf,  SZI);
    k_cvt_bf16<<<eb(SZI), tb, 0, stream>>>(shape_p, shape_bf, SZI);
    k_packT_cvt<<<eb(SZI), tb, 0, stream>>>(x_p, x_bf, DIN, Nn, SZI);

    // ---- x_e, t, s  (M=512, N=2048, K=128)
    gemm(wx_bf, x_bf,     b_x, x_e, 512, Nn, 128, 0, (long long)DIN * Nn, (long long)512 * Nn, 128);
    gemm(wt_bf, time_bf,  b_t, t_f, 512, Nn, 128, 0, (long long)DIN * Nn, (long long)512 * Nn, 128);
    gemm(ws_bf, shape_bf, b_s, s_f, 512, Nn, 128, 0, (long long)DIN * Nn, (long long)512 * Nn, 128);

    // ---- activation packs for the 512-K GEMMs ([b][n][512])
    k_packT_mulsilu<<<eb(SZ), tb, 0, stream>>>(t_f, s_f, ths_bf, C, Nn, SZ);
    k_packT_relu<<<eb(SZ), tb, 0, stream>>>(s_f, rs_bf, C, Nn, SZ);
    k_packT_silu<<<eb(SZ), tb, 0, stream>>>(t_f, st_bf, C, Nn, SZ);

    // ---- mod = W_ada*silu(ths); kv = W_kv*relu(s); mod2 = W_tt*silu(t)
    gemm(wada_bf, ths_bf, b_ada, mod,  1536, Nn, 512, 0, (long long)C * Nn, (long long)1536 * Nn, 512);
    gemm(wkv_bf,  rs_bf,  b_kv,  kv,   1024, Nn, 512, 0, (long long)C * Nn, (long long)1024 * Nn, 512);
    gemm(wtt_bf,  st_bf,  b_tt,  mod2, 1536, Nn, 512, 0, (long long)C * Nn, (long long)1536 * Nn, 512);

    // ---- BN1 stats + x1 modulation (f32 [B,C,N] + bf16 [b][n][C])
    k_bnstats<<<dim3(C), tb, 0, stream>>>(x_e, mean1, var1, Bn, C, Nn);
    k_modulate<<<eb(SZ), tb, 0, stream>>>(x_e, mod, mean1, var1, gamma1, beta1,
                                          x1f, x1_bf, C, Nn, SZ);

    // ---- split kv -> shape_k (A op, [C][N]) / shape_v (B op, [j][n] = [C][N])
    k_split_pack<<<eb(SZ), tb, 0, stream>>>(kv, kk_bf, C, Nn, 0, 1024, SZ);
    k_split_pack<<<eb(SZ), tb, 0, stream>>>(kv, vv_bf, C, Nn, C, 1024, SZ);

    // ---- energy = K * V^T : per-b [512 x 2048] x [2048 x 512]
    gemm(kk_bf, vv_bf, nullptr, energy, 512, 512, Nn,
         (long long)C * Nn, (long long)C * Nn, (long long)C * C, Nn);

    // ---- softmax rows + column renorm, pack to bf16
    k_softmax_row<<<dim3(C, Bn), tb, 0, stream>>>(energy, C);
    k_colsum<<<dim3(C / 256, Bn), tb, 0, stream>>>(energy, colsum, C);
    k_attn_pack<<<eb(SZE), tb, 0, stream>>>(energy, colsum, attn_bf, C, SZE);

    // ---- x_shape = attn * x1 : per-b [512 x 512] x [512 x 2048]
    gemm(attn_bf, x1_bf, nullptr, xshape, 512, Nn, 512,
         (long long)C * C, (long long)Nn * C, (long long)C * Nn, 512);

    // ---- xo = x_e + gate_g * (x1 - x_shape)
    k_xo<<<eb(SZ), tb, 0, stream>>>(x_e, mod, x1f, xshape, xo, C, Nn, SZ);

    // ---- BN2 stats + h modulation (bf16 [b][n][C] only)
    k_bnstats<<<dim3(C), tb, 0, stream>>>(xo, mean2, var2, Bn, C, Nn);
    k_modulate<<<eb(SZ), tb, 0, stream>>>(xo, mod2, mean2, var2, gamma2, beta2,
                                          nullptr, h_bf, C, Nn, SZ);

    // ---- MLP: m1 -> silu -> m2
    gemm(wm1_bf, h_bf, b_m1, m1o, 1024, Nn, 512, 0, (long long)Nn * C, (long long)1024 * Nn, 512);
    k_packT_silu<<<eb(2 * SZ), tb, 0, stream>>>(m1o, m1s_bf, 1024, Nn, 2 * SZ);
    gemm(wm2_bf, m1s_bf, b_m2, mlp, 512, Nn, 1024, 0, (long long)Nn * 1024, (long long)C * Nn, 1024);

    // ---- out = shortcut + gate_l * mlp
    k_final<<<eb(SZ), tb, 0, stream>>>(x_e, mod2, mlp, out, C, Nn, SZ);
}
